// MambaPyTorch_75840532513176
// MI455X (gfx1250) — matrix operand order repaired and checked by reading, not compiled, not run on hardware
//
#include <hip/hip_runtime.h>
#include <hip/hip_bf16.h>
#include <stdint.h>

// ---------------------------------------------------------------------------
// Mamba block forward for MI455X (gfx1250), bf16 WMMA GEMMs + register scan.
// GEMM: 128x256 block tile, 8 waves (2x4), 64x64 wave tile = 16 WMMA / K-step.
// ---------------------------------------------------------------------------

#define D_MODEL 1024
#define D_STATE 16
#define D_CONV  4
#define D_INNER 2048
#define BATCH   2
#define SEQ     2048
#define NTOK    (BATCH * SEQ)           // 4096 tokens
#define DBC_W   (D_INNER + 2 * D_STATE) // 2080

typedef __attribute__((ext_vector_type(16))) __bf16 v16bf;
typedef __attribute__((ext_vector_type(8)))  float  v8f;

union FragBF {           // 16 bf16 = 32B, assembled from two 16B chunks
    v16bf v;
    uint4 q[2];
};

__device__ __forceinline__ unsigned short f32_to_bf16(float f) {
    unsigned int u = __float_as_uint(f);
    unsigned int r = (u >> 16) & 1u;
    u += 0x7FFFu + r;                  // round-to-nearest-even
    return (unsigned short)(u >> 16);
}
__device__ __forceinline__ float silu_f(float x) {
    return x / (1.0f + __expf(-x));
}

// ---------------------------------------------------------------------------
// Generic f32 -> bf16 cast (grid-stride)
// ---------------------------------------------------------------------------
__global__ void cast_f32_bf16_kernel(const float* __restrict__ in,
                                     unsigned short* __restrict__ out,
                                     long long n) {
    long long i = (long long)blockIdx.x * blockDim.x + threadIdx.x;
    long long s = (long long)gridDim.x * blockDim.x;
    for (; i < n; i += s) out[i] = f32_to_bf16(in[i]);
}

// ---------------------------------------------------------------------------
// WMMA GEMM:  C[M,N] = A[M,K] @ B[N,K]^T   (A,B bf16 row-major, C f32)
//   block tile 128x256, BK=32, 256 threads = 8 waves (2 in M x 4 in N),
//   wave tile 64x64 -> 4x4 = 16 v_wmma_f32_16x16x32_bf16 per K-step.
//   EPI==0: plain store.  EPI==1: + bias[n], softplus.
//   M must be a multiple of 128, K a multiple of 32 (true for all uses).
// ---------------------------------------------------------------------------
#define BM 128
#define BN 256
#define BK 32
#define SA 40   // padded LDS row stride (elements)

template <int EPI>
__global__ __launch_bounds__(256)
void gemm_bf16_wmma_kernel(const unsigned short* __restrict__ A,
                           const unsigned short* __restrict__ B,
                           float* __restrict__ C,
                           int M, int N, int K,
                           const float* __restrict__ bias) {
    __shared__ __align__(16) unsigned short As[BM * SA];
    __shared__ __align__(16) unsigned short Bs[BN * SA];

    const int tid  = threadIdx.x;
    const int lane = tid & 31;
    const int wid  = tid >> 5;
    const int wm   = wid >> 2;           // 0..1  (64 rows each)
    const int wn   = wid & 3;            // 0..3  (64 cols each)
    const int blockM = blockIdx.y * BM;
    const int blockN = blockIdx.x * BN;

    const float zf = 0.0f;
    v8f acc[4][4];
#pragma unroll
    for (int i = 0; i < 4; ++i)
#pragma unroll
        for (int j = 0; j < 4; ++j)
            acc[i][j] = (v8f){zf, zf, zf, zf, zf, zf, zf, zf};

    const int lrow = tid >> 2;           // 0..63
    const int lcol = (tid & 3) * 8;      // 0,8,16,24

    for (int k0 = 0; k0 < K; k0 += BK) {
        // ---- stage 128x32 A tile and 256x32 B tile into LDS ----
#pragma unroll
        for (int pass = 0; pass < 2; ++pass) {   // A: 128 rows
            const int row = pass * 64 + lrow;
            const uint4 av = *(const uint4*)(A + (size_t)(blockM + row) * K + k0 + lcol);
            *(uint4*)&As[row * SA + lcol] = av;
        }
#pragma unroll
        for (int pass = 0; pass < 4; ++pass) {   // B: 256 rows, N-guarded
            const int row = pass * 64 + lrow;
            const int gn  = blockN + row;
            uint4 bv;
            if (gn < N) {
                bv = *(const uint4*)(B + (size_t)gn * K + k0 + lcol);
            } else {
                bv.x = bv.y = bv.z = bv.w = 0u;
            }
            *(uint4*)&Bs[row * SA + lcol] = bv;
        }
        // prefetch next K tile into caches (global_prefetch_b8)
        if (k0 + BK < K) {
            __builtin_prefetch(A + (size_t)(blockM + lrow) * K + k0 + BK + lcol, 0, 1);
            __builtin_prefetch(B + (size_t)(blockN + lrow) * K + k0 + BK + lcol, 0, 1);
        }
        __syncthreads();

        // ---- load fragments (ISA 7.12.2 layouts) ----
        FragBF a[4], b[4];
        const int kh = (lane >> 4) * 8;   // A: K half select
        const int kb = (lane >> 4) * 8;   // B: K half select
#pragma unroll
        for (int i = 0; i < 4; ++i) {
            const int r = wm * 64 + i * 16 + (lane & 15);
            a[i].q[0] = *(const uint4*)&As[r * SA + kh];        // K kh..kh+7
            a[i].q[1] = *(const uint4*)&As[r * SA + kh + 16];   // K kh+16..kh+23
        }
#pragma unroll
        for (int j = 0; j < 4; ++j) {
            const int nl = wn * 64 + j * 16 + (lane & 15);
            b[j].q[0] = *(const uint4*)&Bs[nl * SA + kb];       // K kb..kb+7
            b[j].q[1] = *(const uint4*)&Bs[nl * SA + kb + 16];  // K kb+16..kb+23
        }

        // ---- 16 WMMAs: D = A x B + C, f32 accumulate ----
#pragma unroll
        for (int i = 0; i < 4; ++i)
#pragma unroll
            for (int j = 0; j < 4; ++j)
                acc[i][j] = __builtin_amdgcn_wmma_f32_16x16x32_bf16(
                    false, a[i].v, false, b[j].v,
                    (short)0, acc[i][j], false, false);
        __syncthreads();
    }

    // ---- epilogue + store ----
#pragma unroll
    for (int i = 0; i < 4; ++i) {
        const int mbase = blockM + wm * 64 + i * 16 + ((lane >> 4) * 8);
#pragma unroll
        for (int j = 0; j < 4; ++j) {
            const int ncol = blockN + wn * 64 + j * 16 + (lane & 15);
            if (ncol < N) {
#pragma unroll
                for (int v = 0; v < 8; ++v) {
                    float val = acc[i][j][v];
                    if (EPI == 1) {
                        val += bias[ncol];
                        val = (val > 20.0f) ? val : log1pf(__expf(val));
                    }
                    C[(size_t)(mbase + v) * N + ncol] = val;
                }
            }
        }
    }
}

// ---------------------------------------------------------------------------
// Causal depthwise conv (k=4) + bias + SiLU.  Reads x_ssm = xz[:, :2048].
// Writes x_act (f32) and x_act (bf16) for the next GEMM.
// ---------------------------------------------------------------------------
__global__ void conv_silu_kernel(const float* __restrict__ xz,
                                 const float* __restrict__ conv_w,
                                 const float* __restrict__ conv_b,
                                 float* __restrict__ xact,
                                 unsigned short* __restrict__ xact_bf) {
    const long long idx = (long long)blockIdx.x * blockDim.x + threadIdx.x;
    if (idx >= (long long)NTOK * D_INNER) return;
    const int d = (int)(idx & (D_INNER - 1));
    const int m = (int)(idx >> 11);
    const int b = m >> 11;               // SEQ == 2048
    const int t = m & (SEQ - 1);

    float acc = conv_b[d];
#pragma unroll
    for (int j = 0; j < D_CONV; ++j) {
        const int tt = t + j - (D_CONV - 1);
        if (tt >= 0)
            acc += conv_w[d * D_CONV + j] *
                   xz[(size_t)((b << 11) + tt) * (2 * D_INNER) + d];
    }
    const float s = silu_f(acc);
    xact[idx]    = s;
    xact_bf[idx] = f32_to_bf16(s);
}

// ---------------------------------------------------------------------------
// Slice delta_raw = dBC[:, :2048] -> bf16 contiguous (for dt_proj GEMM)
// ---------------------------------------------------------------------------
__global__ void slice_dr_bf16_kernel(const float* __restrict__ dBC,
                                     unsigned short* __restrict__ dr_bf) {
    const long long idx = (long long)blockIdx.x * blockDim.x + threadIdx.x;
    if (idx >= (long long)NTOK * D_INNER) return;
    const int d = (int)(idx & (D_INNER - 1));
    const int m = (int)(idx >> 11);
    dr_bf[idx] = f32_to_bf16(dBC[(size_t)m * DBC_W + d]);
}

// ---------------------------------------------------------------------------
// Selective scan: one lane per (batch, channel). 16-state recurrence in
// registers, fused with  y = (y + u*D) * silu(z), store bf16 for out_proj.
// ---------------------------------------------------------------------------
__global__ __launch_bounds__(64)
void selective_scan_kernel(const float* __restrict__ delta,   // (NTOK, 2048)
                           const float* __restrict__ xact,    // (NTOK, 2048)
                           const float* __restrict__ dBC,     // (NTOK, 2080)
                           const float* __restrict__ xz,      // (NTOK, 4096)
                           const float* __restrict__ A_log,   // (2048, 16)
                           const float* __restrict__ Dvec,    // (2048,)
                           unsigned short* __restrict__ y_bf) {
    const int tid = blockIdx.x * 64 + threadIdx.x;
    if (tid >= BATCH * D_INNER) return;
    const int b = tid >> 11;
    const int d = tid & (D_INNER - 1);

    float An[D_STATE];
#pragma unroll
    for (int n = 0; n < D_STATE; ++n)
        An[n] = -__expf(A_log[d * D_STATE + n]);
    const float Dd = Dvec[d];

    float h[D_STATE];
#pragma unroll
    for (int n = 0; n < D_STATE; ++n) h[n] = 0.0f;

    for (int t = 0; t < SEQ; ++t) {
        const int m = (b << 11) + t;
        const float dt = delta[(size_t)m * D_INNER + d];
        const float u  = xact[(size_t)m * D_INNER + d];
        const float z  = xz[(size_t)m * (2 * D_INNER) + D_INNER + d];

        union { float4 v[4]; float f[16]; } Bv, Cv;
        const float4* bp = (const float4*)(dBC + (size_t)m * DBC_W + D_INNER);
        const float4* cp = (const float4*)(dBC + (size_t)m * DBC_W + D_INNER + D_STATE);
#pragma unroll
        for (int q = 0; q < 4; ++q) { Bv.v[q] = bp[q]; Cv.v[q] = cp[q]; }

        const float du = dt * u;
        float y = 0.0f;
#pragma unroll
        for (int n = 0; n < D_STATE; ++n) {
            const float dA = __expf(dt * An[n]);   // ZOH discretization
            h[n] = dA * h[n] + du * Bv.f[n];
            y += h[n] * Cv.f[n];
        }
        y = (y + u * Dd) * silu_f(z);
        y_bf[(size_t)m * D_INNER + d] = f32_to_bf16(y);
    }
}

// ---------------------------------------------------------------------------
// Host-side launch
// ---------------------------------------------------------------------------
extern "C" void kernel_launch(void* const* d_in, const int* in_sizes, int n_in,
                              void* d_out, int out_size, void* d_ws, size_t ws_size,
                              hipStream_t stream) {
    (void)in_sizes; (void)n_in; (void)out_size; (void)ws_size;
    const float* x         = (const float*)d_in[0];   // (2,2048,1024)
    const float* in_proj_w = (const float*)d_in[1];   // (4096,1024)
    const float* conv_w    = (const float*)d_in[2];   // (2048,1,4)
    const float* conv_b    = (const float*)d_in[3];   // (2048,)
    const float* x_proj_w  = (const float*)d_in[4];   // (2080,2048)
    const float* dt_proj_w = (const float*)d_in[5];   // (2048,2048)
    const float* dt_proj_b = (const float*)d_in[6];   // (2048,)
    const float* A_log     = (const float*)d_in[7];   // (2048,16)
    const float* Dvec      = (const float*)d_in[8];   // (2048,)
    const float* out_proj_w= (const float*)d_in[9];   // (1024,2048)
    float* out = (float*)d_out;                       // (2,2048,1024)

    // ---- workspace layout (bytes) ----
    char* ws = (char*)d_ws;
    size_t off = 0;
    auto take = [&](size_t bytes) { char* p = ws + off; off += (bytes + 255) & ~(size_t)255; return p; };
    unsigned short* w_in_bf  = (unsigned short*)take((size_t)2 * D_INNER * D_MODEL * 2);  // 4096x1024
    unsigned short* w_x_bf   = (unsigned short*)take((size_t)DBC_W * D_INNER * 2);        // 2080x2048
    unsigned short* w_dt_bf  = (unsigned short*)take((size_t)D_INNER * D_INNER * 2);
    unsigned short* w_out_bf = (unsigned short*)take((size_t)D_MODEL * D_INNER * 2);
    unsigned short* x_bf     = (unsigned short*)take((size_t)NTOK * D_MODEL * 2);
    float*          xz       = (float*)take((size_t)NTOK * 2 * D_INNER * 4);              // 4096x4096
    float*          xact     = (float*)take((size_t)NTOK * D_INNER * 4);
    unsigned short* xact_bf  = (unsigned short*)take((size_t)NTOK * D_INNER * 2);
    float*          dBC      = (float*)take((size_t)NTOK * DBC_W * 4);
    unsigned short* dr_bf    = (unsigned short*)take((size_t)NTOK * D_INNER * 2);
    float*          delta    = (float*)take((size_t)NTOK * D_INNER * 4);
    unsigned short* y_bf     = (unsigned short*)take((size_t)NTOK * D_INNER * 2);

    // ---- 1. casts to bf16 ----
    auto cast = [&](const float* src, unsigned short* dst, long long n) {
        int blocks = (int)((n + 255) / 256); if (blocks > 8192) blocks = 8192;
        cast_f32_bf16_kernel<<<blocks, 256, 0, stream>>>(src, dst, n);
    };
    cast(in_proj_w,  w_in_bf,  (long long)2 * D_INNER * D_MODEL);
    cast(x_proj_w,   w_x_bf,   (long long)DBC_W * D_INNER);
    cast(dt_proj_w,  w_dt_bf,  (long long)D_INNER * D_INNER);
    cast(out_proj_w, w_out_bf, (long long)D_MODEL * D_INNER);
    cast(x,          x_bf,     (long long)NTOK * D_MODEL);

    // ---- 2. in_proj: xz = x @ in_proj_w^T   (M=4096,N=4096,K=1024) ----
    {
        dim3 grid((2 * D_INNER + BN - 1) / BN, NTOK / BM);
        gemm_bf16_wmma_kernel<0><<<grid, 256, 0, stream>>>(
            x_bf, w_in_bf, xz, NTOK, 2 * D_INNER, D_MODEL, nullptr);
    }

    // ---- 3. causal conv + SiLU ----
    {
        long long n = (long long)NTOK * D_INNER;
        conv_silu_kernel<<<(int)((n + 255) / 256), 256, 0, stream>>>(
            xz, conv_w, conv_b, xact, xact_bf);
    }

    // ---- 4. x_proj: dBC = x_act @ x_proj_w^T   (N=2080,K=2048) ----
    {
        dim3 grid((DBC_W + BN - 1) / BN, NTOK / BM);
        gemm_bf16_wmma_kernel<0><<<grid, 256, 0, stream>>>(
            xact_bf, w_x_bf, dBC, NTOK, DBC_W, D_INNER, nullptr);
    }

    // ---- 5. slice delta_raw -> bf16 ----
    {
        long long n = (long long)NTOK * D_INNER;
        slice_dr_bf16_kernel<<<(int)((n + 255) / 256), 256, 0, stream>>>(dBC, dr_bf);
    }

    // ---- 6. dt_proj + bias + softplus: delta (N=2048,K=2048) ----
    {
        dim3 grid((D_INNER + BN - 1) / BN, NTOK / BM);
        gemm_bf16_wmma_kernel<1><<<grid, 256, 0, stream>>>(
            dr_bf, w_dt_bf, delta, NTOK, D_INNER, D_INNER, dt_proj_b);
    }

    // ---- 7. selective scan + gating epilogue -> y_bf ----
    selective_scan_kernel<<<(BATCH * D_INNER) / 64, 64, 0, stream>>>(
        delta, xact, dBC, xz, A_log, Dvec, y_bf);

    // ---- 8. out_proj: out = y @ out_proj_w^T   (N=1024,K=2048) ----
    {
        dim3 grid((D_MODEL + BN - 1) / BN, NTOK / BM);
        gemm_bf16_wmma_kernel<0><<<grid, 256, 0, stream>>>(
            y_bf, w_out_bf, out, NTOK, D_MODEL, D_INNER, nullptr);
    }
}
